// CrossAttention_1443109012038
// MI455X (gfx1250) — matrix operand order repaired and checked by reading, not compiled, not run on hardware
//
#include <hip/hip_runtime.h>
#include <hip/hip_bf16.h>

// ---------------------------------------------------------------------------
// CrossAttention forward for MI455X (gfx1250, wave32, WMMA).
//   q = query@Wq^T+bq ; k = key@Wk^T+bk ; v = value@Wv^T+bv
//   attn = softmax(q k^T * scale) ; ctx = attn v ; out = ctx@Wo^T+bo
// B=4, Lq=1024, Lk=2048, D=1024, KD=512, H=16, HD=64
// ---------------------------------------------------------------------------

#define B_   4
#define LQ   1024
#define LK   2048
#define D_   1024
#define KD_  512
#define H_   16
#define HD_  64
#define SCALE_ 0.125f   // 64^-0.5

typedef __attribute__((ext_vector_type(16))) __bf16 v16bf;
typedef __attribute__((ext_vector_type(8)))  float  v8f;

union V16U { v16bf v; uint4 u[2]; };

// ---- gfx1250 async global->LDS path (ASYNCcnt), with sync fallback --------
#if defined(__HIP_DEVICE_COMPILE__) && \
    __has_builtin(__builtin_amdgcn_global_load_async_to_lds_b128) && \
    __has_builtin(__builtin_amdgcn_s_wait_asynccnt)
#define ASYNC_LDS 1
#else
#define ASYNC_LDS 0
#endif

#if ASYNC_LDS
typedef int v4i_t __attribute__((vector_size(16)));
typedef __attribute__((address_space(1))) v4i_t as1_v4i;
typedef __attribute__((address_space(3))) v4i_t as3_v4i;
__device__ __forceinline__ void cp16_g2l(void* lds_dst, const void* gsrc) {
    __builtin_amdgcn_global_load_async_to_lds_b128(
        (as1_v4i*)gsrc, (as3_v4i*)lds_dst, 0, 0);
}
__device__ __forceinline__ void wait_stage0() {
    __builtin_amdgcn_s_wait_asynccnt(0);
}
#else
__device__ __forceinline__ void cp16_g2l(void* lds_dst, const void* gsrc) {
    *reinterpret_cast<uint4*>(lds_dst) = *reinterpret_cast<const uint4*>(gsrc);
}
__device__ __forceinline__ void wait_stage0() {}
#endif

// --------------------------- f32 -> bf16 cast ------------------------------
__global__ __launch_bounds__(256) void cast_f32_bf16_x4(
    const float4* __restrict__ in, __bf16* __restrict__ out, int n4)
{
    int i = blockIdx.x * 256 + threadIdx.x;
    if (i >= n4) return;
    float4 f = in[i];
    union { __bf16 h[4]; uint2 u; } p;
    p.h[0] = (__bf16)f.x; p.h[1] = (__bf16)f.y;
    p.h[2] = (__bf16)f.z; p.h[3] = (__bf16)f.w;
    *reinterpret_cast<uint2*>(out + (size_t)i * 4) = p.u;
}

// --------------------------- GEMM: X @ W^T + b -----------------------------
// A: [M,Kd] bf16 row-major. W: [N,Kd] bf16 row-major. 128x64 tile per WG.
// 8 waves x (16M x 64N); K stepped by 32 via v_wmma_f32_16x16x32_bf16.
// W tiles double-buffered in LDS, filled by async global->LDS DMA.
__global__ __launch_bounds__(256) void gemm_bf16_wmma(
    const __bf16* __restrict__ A, const __bf16* __restrict__ W,
    const float* __restrict__ bias,
    float* __restrict__ out_f32, __bf16* __restrict__ out_bf16,
    __bf16* __restrict__ out_bf16T,
    int M, int N, int Kd)
{
    const int m0   = blockIdx.x * 128;
    const int n0   = blockIdx.y * 64;
    const int tid  = threadIdx.x;
    const int wave = tid >> 5;
    const int lane = tid & 31;
    const int half = lane >> 4;   // C rows 0-7 vs 8-15 ; A/B K-halves
    const int lc   = lane & 15;

    __shared__ __align__(16) __bf16 ldsW[2][64 * 32];   // 2 x 4 KB W chunks

    v8f acc[4];
    for (int t = 0; t < 4; ++t)
        acc[t] = (v8f){0.f,0.f,0.f,0.f,0.f,0.f,0.f,0.f};

    const __bf16* arow = A + (size_t)(m0 + wave * 16 + lc) * Kd;

    // one 16B async copy per thread fills a 64x32 bf16 tile (4 KB)
    auto stage_w = [&](int buf, int k0) {
        int r = tid >> 2, q = tid & 3;
        cp16_g2l(&ldsW[buf][r * 32 + q * 8],
                 W + (size_t)(n0 + r) * Kd + k0 + q * 8);
    };

    stage_w(0, 0);
    for (int k0 = 0; k0 < Kd; k0 += 32) {
        const int  cur  = (k0 >> 5) & 1;
        const bool more = (k0 + 32) < Kd;
        if (more) {
            stage_w(cur ^ 1, k0 + 32);        // fill next buffer in flight
#if ASYNC_LDS
            __builtin_amdgcn_s_wait_asynccnt(1);  // cur tile landed
#endif
        } else {
            wait_stage0();
        }
        __syncthreads();

        // A operand, ISA 16-bit 16x32 layout: lane m holds K {0..7,16..23},
        // lane m+16 holds K {8..15,24..31}
        V16U a;
        a.u[0] = *reinterpret_cast<const uint4*>(arow + k0 + (half ? 8 : 0));
        a.u[1] = *reinterpret_cast<const uint4*>(arow + k0 + 16 + (half ? 8 : 0));

        const __bf16* wb = ldsW[cur];
        for (int t = 0; t < 4; ++t) {
            // B operand 32x16: lane n = N; lanes 0-15 K={0..7,16..23}, 16-31 K={8..15,24..31}
            V16U b;
            const __bf16* bp = &wb[(t * 16 + lc) * 32 + (half ? 8 : 0)];
            b.u[0] = *reinterpret_cast<const uint4*>(bp);
            b.u[1] = *reinterpret_cast<const uint4*>(bp + 16);
            acc[t] = __builtin_amdgcn_wmma_f32_16x16x32_bf16(
                false, a.v, false, b.v, (short)0, acc[t], false, false);
        }
        __syncthreads();   // everyone done with cur before it is refilled
    }

    const int rbase = m0 + wave * 16 + half * 8;
    for (int t = 0; t < 4; ++t) {
        const int col = n0 + t * 16 + lc;
        const float bv = bias ? bias[col] : 0.0f;
        if (out_f32) {
            for (int r = 0; r < 8; ++r)
                out_f32[(size_t)(rbase + r) * N + col] = acc[t][r] + bv;
        }
        if (out_bf16) {
            for (int r = 0; r < 8; ++r)
                out_bf16[(size_t)(rbase + r) * N + col] = (__bf16)(acc[t][r] + bv);
        }
        if (out_bf16T) {  // [N, M]: lane's 8 consecutive M values -> one b128 store
            union { __bf16 h[8]; uint4 u; } pk;
            for (int r = 0; r < 8; ++r) pk.h[r] = (__bf16)(acc[t][r] + bv);
            *reinterpret_cast<uint4*>(&out_bf16T[(size_t)col * M + rbase]) = pk.u;
        }
    }
}

// ------------------------------ attention ----------------------------------
// One WG = one (b,h) x 128 queries. 8 waves x 16 query rows.
// K / V^T tiles async-staged to LDS once per 128-key chunk and shared by all
// 8 waves as WMMA B-operands. Pass 1: online softmax stats. Pass 2: recompute
// scores, stream normalized attn with NT stores, transpose P via per-wave
// LDS, WMMA against staged V^T for context.
__global__ __launch_bounds__(256) void attn_wmma(
    const __bf16* __restrict__ Q,    // [B*Lq, D] row-major
    const __bf16* __restrict__ K,    // [B*Lk, D] row-major
    const __bf16* __restrict__ VT,   // [D, B*Lk] (transposed v projection)
    float* __restrict__ attn,        // [B,H,Lq,Lk]
    __bf16* __restrict__ ctx)        // [B*Lq, D]
{
    const int qb = blockIdx.x;               // Lq/128
    const int bh = blockIdx.y;               // B*H
    const int b = bh >> 4, h = bh & 15;
    const int tid = threadIdx.x, wave = tid >> 5, lane = tid & 31;
    const int half = lane >> 4, lc = lane & 15;

    __shared__ __align__(16) __bf16 kt_lds[128 * 64];    // [key][hd]   16 KB
    __shared__ __align__(16) __bf16 vt_lds[64 * 128];    // [hd][key]   16 KB
    __shared__ __align__(16) __bf16 pls[8][16][128];     // per-wave P  32 KB

    const int q0 = qb * 128 + wave * 16;
    const __bf16* qrow = Q + (size_t)(b * LQ + q0 + lc) * D_ + h * HD_;

    V16U qa[2];                               // q 16x64 = two 16x32 A chunks
    for (int c = 0; c < 2; ++c) {
        qa[c].u[0] = *reinterpret_cast<const uint4*>(qrow + c * 32 + (half ? 8 : 0));
        qa[c].u[1] = *reinterpret_cast<const uint4*>(qrow + c * 32 + 16 + (half ? 8 : 0));
    }

    const char* kbase_b = (const char*)(K + (size_t)(b * LK) * D_ + h * HD_);
    const char* vt_b    = (const char*)(VT + (size_t)(h * HD_) * (B_ * LK) + b * LK);

    // 128x64 bf16 K chunk: flat 16 KB, 4 x 16B per thread
    auto stage_k = [&](int kt) {
        for (int j = 0; j < 4; ++j) {
            int o = (tid + j * 256) * 16;         // byte offset in tile
            int row = o >> 7, colb = o & 127;     // 128 B per key row
            cp16_g2l((char*)kt_lds + o,
                     kbase_b + (size_t)(kt + row) * (D_ * 2) + colb);
        }
    };
    // 64x128 bf16 V^T chunk: flat 16 KB, 4 x 16B per thread
    auto stage_v = [&](int kt) {
        for (int j = 0; j < 4; ++j) {
            int o = (tid + j * 256) * 16;
            int row = o >> 8, colb = o & 255;     // 256 B per hd row
            cp16_g2l((char*)vt_lds + o,
                     vt_b + (size_t)row * (B_ * LK * 2) + (size_t)kt * 2 + colb);
        }
    };

    float mrow[8], lrow[8];
    for (int r = 0; r < 8; ++r) { mrow[r] = -1.0e30f; lrow[r] = 0.0f; }

    // ---------------- pass 1: online softmax stats ----------------
    for (int kt = 0; kt < LK; kt += 128) {
        stage_k(kt);
        wait_stage0();
        __syncthreads();

        v8f s[8];
        for (int t = 0; t < 8; ++t) {
            s[t] = (v8f){0.f,0.f,0.f,0.f,0.f,0.f,0.f,0.f};
            const __bf16* krow = &kt_lds[(t * 16 + lc) * 64];
            for (int c = 0; c < 2; ++c) {
                V16U kb;
                const __bf16* kp = krow + c * 32 + (half ? 8 : 0);
                kb.u[0] = *reinterpret_cast<const uint4*>(kp);
                kb.u[1] = *reinterpret_cast<const uint4*>(kp + 16);
                s[t] = __builtin_amdgcn_wmma_f32_16x16x32_bf16(
                    false, qa[c].v, false, kb.v, (short)0, s[t], false, false);
            }
        }
        __syncthreads();   // kt_lds consumed; safe to refill next iteration

        for (int t = 0; t < 8; ++t)
            for (int r = 0; r < 8; ++r) s[t][r] *= SCALE_;
        for (int r = 0; r < 8; ++r) {
            float cm = s[0][r];
            for (int t = 1; t < 8; ++t) cm = fmaxf(cm, s[t][r]);
            cm = fmaxf(cm, __shfl_xor(cm, 1, 16));
            cm = fmaxf(cm, __shfl_xor(cm, 2, 16));
            cm = fmaxf(cm, __shfl_xor(cm, 4, 16));
            cm = fmaxf(cm, __shfl_xor(cm, 8, 16));
            const float mnew = fmaxf(mrow[r], cm);
            float ps = 0.0f;
            for (int t = 0; t < 8; ++t) ps += __expf(s[t][r] - mnew);
            ps += __shfl_xor(ps, 1, 16);
            ps += __shfl_xor(ps, 2, 16);
            ps += __shfl_xor(ps, 4, 16);
            ps += __shfl_xor(ps, 8, 16);
            lrow[r] = lrow[r] * __expf(mrow[r] - mnew) + ps;
            mrow[r] = mnew;
        }
    }
    float inv_l[8];
    for (int r = 0; r < 8; ++r) inv_l[r] = 1.0f / lrow[r];

    // ------- pass 2: recompute, emit attn (NT), accumulate context ---------
    v8f cacc[4];
    for (int t = 0; t < 4; ++t)
        cacc[t] = (v8f){0.f,0.f,0.f,0.f,0.f,0.f,0.f,0.f};

    float* arow_out = attn + ((size_t)bh * LQ + q0) * LK;

    for (int kt = 0; kt < LK; kt += 128) {
        stage_k(kt);
        stage_v(kt);
        wait_stage0();
        __syncthreads();

        v8f s[8];
        for (int t = 0; t < 8; ++t) {
            s[t] = (v8f){0.f,0.f,0.f,0.f,0.f,0.f,0.f,0.f};
            const __bf16* krow = &kt_lds[(t * 16 + lc) * 64];
            for (int c = 0; c < 2; ++c) {
                V16U kb;
                const __bf16* kp = krow + c * 32 + (half ? 8 : 0);
                kb.u[0] = *reinterpret_cast<const uint4*>(kp);
                kb.u[1] = *reinterpret_cast<const uint4*>(kp + 16);
                s[t] = __builtin_amdgcn_wmma_f32_16x16x32_bf16(
                    false, qa[c].v, false, kb.v, (short)0, s[t], false, false);
            }
        }

        // normalize; stream attn f32 (non-temporal, 512 MB single-use);
        // stage P (bf16) into per-wave LDS for A-operand transpose
        for (int t = 0; t < 8; ++t) {
            for (int r = 0; r < 8; ++r) {
                const int row = half * 8 + r;
                const float p = __expf(s[t][r] * SCALE_ - mrow[r]) * inv_l[r];
                __builtin_nontemporal_store(
                    p, &arow_out[(size_t)row * LK + kt + t * 16 + lc]);
                pls[wave][row][t * 16 + lc] = (__bf16)p;
            }
        }

        // context += P(16x128) @ V(128x64) from staged V^T
        for (int c = 0; c < 4; ++c) {
            V16U pa;
            const __bf16* pp = &pls[wave][lc][c * 32];
            pa.u[0] = *reinterpret_cast<const uint4*>(pp + (half ? 8 : 0));
            pa.u[1] = *reinterpret_cast<const uint4*>(pp + 16 + (half ? 8 : 0));
            for (int t2 = 0; t2 < 4; ++t2) {
                V16U vb;
                const __bf16* vp = &vt_lds[(t2 * 16 + lc) * 128 + c * 32 + (half ? 8 : 0)];
                vb.u[0] = *reinterpret_cast<const uint4*>(vp);
                vb.u[1] = *reinterpret_cast<const uint4*>(vp + 16);
                cacc[t2] = __builtin_amdgcn_wmma_f32_16x16x32_bf16(
                    false, pa.v, false, vb.v, (short)0, cacc[t2], false, false);
            }
        }
        __syncthreads();   // kt_lds / vt_lds consumed
    }

    // write context (bf16) into [B*Lq, D] at head column block
    const int rbase = q0 + half * 8;
    for (int t2 = 0; t2 < 4; ++t2) {
        const int col = h * HD_ + t2 * 16 + lc;
        for (int r = 0; r < 8; ++r)
            ctx[(size_t)(b * LQ + rbase + r) * D_ + col] = (__bf16)cacc[t2][r];
    }
}

// ------------------------------ launcher -----------------------------------
extern "C" void kernel_launch(void* const* d_in, const int* in_sizes, int n_in,
                              void* d_out, int out_size, void* d_ws, size_t ws_size,
                              hipStream_t stream) {
    const float* query = (const float*)d_in[0];
    const float* key   = (const float*)d_in[1];
    const float* value = (const float*)d_in[2];
    const float* Wq = (const float*)d_in[3];
    const float* bq = (const float*)d_in[4];
    const float* Wk = (const float*)d_in[5];
    const float* bk = (const float*)d_in[6];
    const float* Wv = (const float*)d_in[7];
    const float* bv = (const float*)d_in[8];
    const float* Wo = (const float*)d_in[9];
    const float* bo = (const float*)d_in[10];

    float* out  = (float*)d_out;                          // [B,Lq,D]
    float* attn = out + (size_t)B_ * LQ * D_;             // [B,H,Lq,Lk]

    char* ws = (char*)d_ws;
    size_t off = 0;
    auto carve = [&](size_t bytes) {
        void* p = ws + off;
        off += (bytes + 255) & ~(size_t)255;
        return p;
    };
    __bf16* query_b = (__bf16*)carve((size_t)B_ * LQ * D_  * 2);
    __bf16* key_b   = (__bf16*)carve((size_t)B_ * LK * KD_ * 2);
    __bf16* value_b = (__bf16*)carve((size_t)B_ * LK * KD_ * 2);
    __bf16* Wq_b    = (__bf16*)carve((size_t)D_ * D_  * 2);
    __bf16* Wk_b    = (__bf16*)carve((size_t)D_ * KD_ * 2);
    __bf16* Wv_b    = (__bf16*)carve((size_t)D_ * KD_ * 2);
    __bf16* Wo_b    = (__bf16*)carve((size_t)D_ * D_  * 2);
    __bf16* qproj   = (__bf16*)carve((size_t)B_ * LQ * D_ * 2);  // [B*Lq, D]
    __bf16* kproj   = (__bf16*)carve((size_t)B_ * LK * D_ * 2);  // [B*Lk, D]
    __bf16* vprojT  = (__bf16*)carve((size_t)D_ * B_ * LK * 2);  // [D, B*Lk]
    __bf16* ctx_b   = (__bf16*)carve((size_t)B_ * LQ * D_ * 2);  // [B*Lq, D]

    auto cast = [&](const float* src, __bf16* dst, size_t n) {
        int n4 = (int)(n / 4);
        cast_f32_bf16_x4<<<(n4 + 255) / 256, 256, 0, stream>>>(
            (const float4*)src, dst, n4);
    };
    cast(query, query_b, (size_t)B_ * LQ * D_);
    cast(key,   key_b,   (size_t)B_ * LK * KD_);
    cast(value, value_b, (size_t)B_ * LK * KD_);
    cast(Wq, Wq_b, (size_t)D_ * D_);
    cast(Wk, Wk_b, (size_t)D_ * KD_);
    cast(Wv, Wv_b, (size_t)D_ * KD_);
    cast(Wo, Wo_b, (size_t)D_ * D_);

    // q = query@Wq^T+bq  -> bf16 row-major
    gemm_bf16_wmma<<<dim3((B_ * LQ) / 128, D_ / 64), 256, 0, stream>>>(
        query_b, Wq_b, bq, nullptr, qproj, nullptr, B_ * LQ, D_, D_);
    // k = key@Wk^T+bk    -> bf16 row-major
    gemm_bf16_wmma<<<dim3((B_ * LK) / 128, D_ / 64), 256, 0, stream>>>(
        key_b, Wk_b, bk, nullptr, kproj, nullptr, B_ * LK, D_, KD_);
    // v = value@Wv^T+bv  -> bf16 TRANSPOSED [D, B*Lk] for PV B-operand
    gemm_bf16_wmma<<<dim3((B_ * LK) / 128, D_ / 64), 256, 0, stream>>>(
        value_b, Wv_b, bv, nullptr, nullptr, vprojT, B_ * LK, D_, KD_);

    // attention: softmax + attn emit + context
    attn_wmma<<<dim3(LQ / 128, B_ * H_), 256, 0, stream>>>(
        qproj, kproj, vprojT, attn, ctx_b);

    // out = ctx@Wo^T+bo -> f32
    gemm_bf16_wmma<<<dim3((B_ * LQ) / 128, D_ / 64), 256, 0, stream>>>(
        ctx_b, Wo_b, bo, out, nullptr, nullptr, B_ * LQ, D_, D_);
}
